// NeRFRenderer_46703474377280
// MI455X (gfx1250) — matrix operand order repaired and checked by reading, not compiled, not run on hardware
//
#include <hip/hip_runtime.h>
#include <hip/hip_bf16.h>
#include <stdint.h>

typedef _Float16 v16h __attribute__((ext_vector_type(16)));
typedef float    v8f  __attribute__((ext_vector_type(8)));
typedef float    f4   __attribute__((ext_vector_type(4)));
typedef unsigned int v4u __attribute__((ext_vector_type(4)));
typedef int      v8i  __attribute__((ext_vector_type(8)));
typedef int      v4i  __attribute__((ext_vector_type(4)));

union Frag16 { v16h h; f4 f[2]; };

// --------------------------------------------------------------------------
// Tensor Data Mover availability (device pass only)
// --------------------------------------------------------------------------
#if defined(__AMDGCN__) && __has_builtin(__builtin_amdgcn_tensor_load_to_lds) && \
    __has_builtin(__builtin_amdgcn_s_wait_tensorcnt)
#define USE_TDM 1
#else
#define USE_TDM 0
#endif

#if __has_include(<hip/amd_detail/amd_gfx1250_TDM.h>)
#define TDM_SIX_ARGS 1
#else
#define TDM_SIX_ARGS 0
#endif

#if USE_TDM
// Issue one TDM 2D tile load: rows x 64 f16 elements, global row stride = Kpad
// elements, LDS rows padded 64 -> 72 halves via the TDM pad feature
// (pad_interval = 32 dwords -> code 4, pad_amount = 4 dwords -> code 3).
__device__ inline void tdm_issue(const _Float16* gsrc, uint32_t lds_off,
                                 int rows, int kpad) {
    uint64_t ga = (uint64_t)(uintptr_t)gsrc;
    union { uint32_t w[4]; v4u v; } g0;
    g0.w[0] = 1u;                                   // count=1, user descriptor
    g0.w[1] = lds_off;                              // lds_addr (bytes)
    g0.w[2] = (uint32_t)(ga & 0xFFFFFFFFu);         // global_addr[31:0]
    g0.w[3] = (uint32_t)((ga >> 32) & 0x1FFFFFFu)   // global_addr[56:32]
            | (2u << 30);                           // type = 2 ("image")
    union { uint32_t w[8]; v8i v; } g1;
    g1.w[0] = (1u << 16)      // data_size = 1 -> 2 bytes
            | (1u << 20)      // pad_enable
            | (4u << 22)      // pad_interval: 32 dwords (one 128B row)
            | (3u << 25);     // pad_amount: 4 dwords (8 halves)
    g1.w[1] = ((uint32_t)64 << 16);                 // tensor_dim0 = 64 (lo16)
    g1.w[2] = ((uint32_t)(rows & 0xFFFF) << 16);    // tensor_dim1 = rows (lo16)
    g1.w[3] = ((uint32_t)64 << 16);                 // tile_dim0 = 64
    g1.w[4] = (uint32_t)(rows & 0xFFFF);            // tile_dim1 = rows, tile_dim2 = 0
    g1.w[5] = (uint32_t)kpad;                       // tensor_dim0_stride[31:0]
    g1.w[6] = 0;                                    // stride hi, dim1_stride lo
    g1.w[7] = 0;
    union { uint32_t w[4]; v4i v; } g2; g2.w[0] = g2.w[1] = g2.w[2] = g2.w[3] = 0;
    union { uint32_t w[4]; v4i v; } g3; g3.w[0] = g3.w[1] = g3.w[2] = g3.w[3] = 0;
#if TDM_SIX_ARGS
    union { uint32_t w[8]; v8i v; } g4; for (int i = 0; i < 8; ++i) g4.w[i] = 0;
    __builtin_amdgcn_tensor_load_to_lds(g0.v, g1.v, g2.v, g3.v, g4.v, 0);
#else
    __builtin_amdgcn_tensor_load_to_lds(g0.v, g1.v, g2.v, g3.v, 0);
#endif
}
#endif // USE_TDM

// ---------------------------------------------------------------------------
// Weight prep: fp32 (K,N) row-major  ->  f16 W^T (Npad, Kpad) row-major, zero pad
// ---------------------------------------------------------------------------
__global__ void k_prep_weight(const float* __restrict__ W, _Float16* __restrict__ WT,
                              int K, int N, int Kpad, int Npad) {
    int idx = blockIdx.x * blockDim.x + threadIdx.x;
    int total = Kpad * Npad;
    if (idx >= total) return;
    int n = idx / Kpad;
    int k = idx - n * Kpad;
    float v = (k < K && n < N) ? W[(size_t)k * N + n] : 0.0f;
    WT[idx] = (_Float16)v;
}

// ---------------------------------------------------------------------------
// WMMA GEMM: C[M x Npad] = act(A[M x Kpad] * W + bias), f16 in/out, f32 acc.
// One wave -> 16 x (16*NTILES) strip; 4 waves/block over M.
// B panel staged in LDS in 64-K-element chunks (TDM DMA, double buffered),
// rows padded to 72 halves (bank-conflict-free B-frag reads).
// A-frags software-pipelined from global; ALL B-frags of a k-step preloaded
// into distinct register fragments so ds_load latency staggers across WMMAs.
// ---------------------------------------------------------------------------
template <int NTILES, bool RELU>
__global__ __launch_bounds__(128)
void gemm_wmma(const _Float16* __restrict__ A, const _Float16* __restrict__ WT,
               const float* __restrict__ bias, _Float16* __restrict__ C,
               int Kpad, int Npad, int Nreal) {
    constexpr int ROWS = NTILES * 16;
    constexpr int LDSROW = 72;   // 64 payload + 8 pad halves
    __shared__ __align__(16) _Float16 ldsb[2][ROWS * LDSROW];

    const int tid  = threadIdx.x;
    const int lane = tid & 31;
    const int wave = tid >> 5;
    const int m0   = (blockIdx.x * 4 + wave) * 16;
    const int n0   = blockIdx.y * ROWS;
    const int l15  = lane & 15;
    const int sel  = lane >> 4;

    const _Float16* arow = A + (size_t)(m0 + l15) * Kpad;
    const int nchunks  = Kpad >> 6;   // 64 K-elements per chunk
    const int ks_total = Kpad >> 5;   // 32 K-elements per wmma step

    v8f acc[NTILES] = {};

#if USE_TDM
    if (tid == 0) {
        tdm_issue(WT + (size_t)n0 * Kpad, (uint32_t)(uintptr_t)&ldsb[0][0], ROWS, Kpad);
    }
#endif

    // A-frag pipeline: preload k-step 0
    Frag16 aNext;
    aNext.f[0] = *reinterpret_cast<const f4*>(arow + sel * 8);
    aNext.f[1] = *reinterpret_cast<const f4*>(arow + 16 + sel * 8);

    for (int c = 0; c < nchunks; ++c) {
        const int buf = c & 1;
#if USE_TDM
        if (tid == 0) {
            if (c + 1 < nchunks) {
                tdm_issue(WT + (size_t)n0 * Kpad + (c + 1) * 64,
                          (uint32_t)(uintptr_t)&ldsb[(c + 1) & 1][0], ROWS, Kpad);
                __builtin_amdgcn_s_wait_tensorcnt(1);  // chunk c complete
            } else {
                __builtin_amdgcn_s_wait_tensorcnt(0);
            }
        }
        __syncthreads();
#else
        // synchronous cooperative copy
        for (int row = tid; row < ROWS; row += 128) {
            const f4* src = reinterpret_cast<const f4*>(WT + (size_t)(n0 + row) * Kpad + c * 64);
            f4* dst = reinterpret_cast<f4*>(&ldsb[buf][row * LDSROW]);
#pragma unroll
            for (int part = 0; part < 8; ++part) dst[part] = src[part];
        }
        __syncthreads();
#endif

#pragma unroll
        for (int kk = 0; kk < 2; ++kk) {
            Frag16 a = aNext;
            int ksNext = c * 2 + kk + 1;
            if (ksNext < ks_total) {
                aNext.f[0] = *reinterpret_cast<const f4*>(arow + ksNext * 32 + sel * 8);
                aNext.f[1] = *reinterpret_cast<const f4*>(arow + ksNext * 32 + 16 + sel * 8);
            }
            // Preload every B fragment of this k-step into its own registers:
            // the 2*NTILES ds_load_b128 issue as one clause and the WMMA chain
            // consumes them with staggered (mostly hidden) dscnt waits.
            const _Float16* bbase = &ldsb[buf][l15 * LDSROW + kk * 32 + sel * 8];
            Frag16 bfr[NTILES];
#pragma unroll
            for (int j = 0; j < NTILES; ++j) {
                const _Float16* bc = bbase + j * 16 * LDSROW;
                bfr[j].f[0] = *reinterpret_cast<const f4*>(bc);
                bfr[j].f[1] = *reinterpret_cast<const f4*>(bc + 16);
            }
#pragma unroll
            for (int j = 0; j < NTILES; ++j) {
                acc[j] = __builtin_amdgcn_wmma_f32_16x16x32_f16(
                    false, a.h, false, bfr[j].h, (short)0, acc[j], false, false);
            }
        }
        __syncthreads();
    }

#pragma unroll
    for (int j = 0; j < NTILES; ++j) {
        int col = n0 + j * 16 + l15;
        float bv = (col < Nreal) ? bias[col] : 0.0f;
#pragma unroll
        for (int r = 0; r < 8; ++r) {
            int row = m0 + r + sel * 8;
            float v = acc[j][r] + bv;
            if (RELU) v = fmaxf(v, 0.0f);
            C[(size_t)row * Npad + col] = (_Float16)v;
        }
    }
}

// ---------------------------------------------------------------------------
// Coarse steps + bounds (one thread per ray)
// ---------------------------------------------------------------------------
__global__ void k_steps_coarse(const float* __restrict__ noise, float* __restrict__ steps,
                               float* __restrict__ bound, int B) {
    int ray = blockIdx.x * blockDim.x + threadIdx.x;
    if (ray >= B) return;
    const float* nz = noise + (size_t)ray * 64;
    float* st = steps + (size_t)ray * 64;
    float* bd = bound + (size_t)ray * 65;
    float prev = (0.5f) / 64.0f + (nz[0] - 0.5f) / 64.0f;
    st[0] = prev;
    bd[0] = 0.0f;
    for (int s = 1; s < 64; ++s) {
        float cur = (s + 0.5f) / 64.0f + (nz[s] - 0.5f) / 64.0f;
        st[s] = cur;
        bd[s] = 0.5f * (prev + cur);
        prev = cur;
    }
    bd[64] = 1.0f;
}

// ---------------------------------------------------------------------------
// Positional encoding of points -> padded K=64 f16 rows
// ---------------------------------------------------------------------------
__global__ void k_embed_geo(const float* __restrict__ coords, const float* __restrict__ steps,
                            _Float16* __restrict__ X, int n, int M) {
    int p = blockIdx.x * blockDim.x + threadIdx.x;
    if (p >= M) return;
    int ray = p / n;
    int s = p - ray * n;
    float c0 = coords[ray * 4 + 0], c1 = coords[ray * 4 + 1];
    float c2 = coords[ray * 4 + 2], c3 = coords[ray * 4 + 3];
    float t = steps[(size_t)ray * n + s];
    float pt[3];
    pt[0] = c0 * (1.0f - t) + c2 * t;
    pt[1] = c1 * (1.0f - t) + c3 * t;
    pt[2] = t;
    _Float16* row = X + (size_t)p * 64;
    row[0] = (_Float16)pt[0];
    row[1] = (_Float16)pt[1];
    row[2] = (_Float16)pt[2];
    float f = 1.0f;
    for (int fr = 0; fr < 10; ++fr) {
        for (int d = 0; d < 3; ++d) {
            float v = pt[d] * f;
            row[3 + fr * 3 + d]  = (_Float16)__sinf(v);
            row[33 + fr * 3 + d] = (_Float16)__cosf(v);
        }
        f *= 2.0f;
    }
    row[63] = (_Float16)0.0f;
}

// ---------------------------------------------------------------------------
// Density head: relu(geo_feat . dW + db + 0.1)
// ---------------------------------------------------------------------------
__global__ void k_density(const _Float16* __restrict__ feat, const float* __restrict__ dW,
                          const float* __restrict__ db, float* __restrict__ dens, int M) {
    int p = blockIdx.x * blockDim.x + threadIdx.x;
    if (p >= M) return;
    const _Float16* g = feat + (size_t)p * 512;
    float acc = 0.0f;
    for (int i = 0; i < 256; ++i) acc += (float)g[i] * dW[i];
    dens[p] = fmaxf(acc + db[0] + 0.1f, 0.0f);
}

// ---------------------------------------------------------------------------
// Per-ray transmittance / weights (sequential cumsum, one thread per ray)
// ---------------------------------------------------------------------------
__global__ void k_weights(const float* __restrict__ dens, const float* __restrict__ bound,
                          float* __restrict__ wbuf, int n, int B) {
    int ray = blockIdx.x * blockDim.x + threadIdx.x;
    if (ray >= B) return;
    const float* bd = bound + (size_t)ray * (n + 1);
    float cum = 0.0f, Tprev = 1.0f;
    for (int s = 0; s < n; ++s) {
        float dd = (bd[s + 1] - bd[s]) * dens[(size_t)ray * n + s];
        float alpha = 1.0f - __expf(-dd);
        wbuf[(size_t)ray * n + s] = Tprev * alpha;
        cum += dd;
        Tprev = __expf(-cum);
    }
}

// ---------------------------------------------------------------------------
// sample_pdf + merge + bitonic sort: one block (64 threads) per ray
// ---------------------------------------------------------------------------
__global__ void k_sample_sort(const float* __restrict__ steps_c, const float* __restrict__ bound_c,
                              const float* __restrict__ wbuf, const float* __restrict__ u_fine,
                              float* __restrict__ steps_f, float* __restrict__ bound_f) {
    __shared__ float cdf[65];
    __shared__ float merged[128];
    int ray = blockIdx.x;
    int t = threadIdx.x;

    if (t == 0) {
        float tot = 0.0f;
        for (int i = 0; i < 64; ++i) tot += wbuf[(size_t)ray * 64 + i] + 1e-5f;
        float inv = 1.0f / tot;
        cdf[0] = 0.0f;
        float c = 0.0f;
        for (int i = 0; i < 64; ++i) {
            c += (wbuf[(size_t)ray * 64 + i] + 1e-5f) * inv;
            cdf[i + 1] = c;
        }
    }
    __syncthreads();

    {
        float u = u_fine[(size_t)ray * 64 + t];
        int lo = 0, hi = 65;
        while (lo < hi) {
            int mid = (lo + hi) >> 1;
            if (u >= cdf[mid]) lo = mid + 1; else hi = mid;
        }
        int inds = lo;
        int below = inds - 1; if (below < 0) below = 0;
        int above = inds;     if (above > 64) above = 64;
        float cb = cdf[below], ca = cdf[above];
        float bb = bound_c[(size_t)ray * 65 + below];
        float ba = bound_c[(size_t)ray * 65 + above];
        float denom = ca - cb;
        if (denom < 1e-5f) denom = 1.0f;
        float tt = (u - cb) / denom;
        merged[64 + t] = bb + tt * (ba - bb);
        merged[t] = steps_c[(size_t)ray * 64 + t];
    }

    for (int k = 2; k <= 128; k <<= 1) {
        for (int j = k >> 1; j > 0; j >>= 1) {
            __syncthreads();
            int i = ((t & ~(j - 1)) << 1) | (t & (j - 1));
            int p = i | j;
            bool up = ((i & k) == 0);
            float a = merged[i], b = merged[p];
            if (up ? (a > b) : (a < b)) { merged[i] = b; merged[p] = a; }
        }
    }
    __syncthreads();

    for (int q = 0; q < 2; ++q) {
        int i = t + q * 64;
        steps_f[(size_t)ray * 128 + i] = merged[i];
        if (i < 127) bound_f[(size_t)ray * 129 + i + 1] = 0.5f * (merged[i] + merged[i + 1]);
    }
    if (t == 0) {
        bound_f[(size_t)ray * 129 + 0] = 0.0f;
        bound_f[(size_t)ray * 129 + 128] = 1.0f;
    }
}

// ---------------------------------------------------------------------------
// Assemble color-MLP input: [app_feat(256) | dir embed(27) | pad] -> 320
// ---------------------------------------------------------------------------
__global__ void k_embed_col(const _Float16* __restrict__ feat, const float* __restrict__ coords,
                            _Float16* __restrict__ X, int n, int M) {
    int p = blockIdx.x * blockDim.x + threadIdx.x;
    if (p >= M) return;
    int ray = p / n;
    _Float16* row = X + (size_t)p * 320;
    const f4* src = reinterpret_cast<const f4*>(feat + (size_t)p * 512 + 256);
    f4* dst = reinterpret_cast<f4*>(row);
#pragma unroll 8
    for (int i = 0; i < 32; ++i) dst[i] = src[i];

    float d0 = coords[ray * 4 + 2] - coords[ray * 4 + 0];
    float d1 = coords[ray * 4 + 3] - coords[ray * 4 + 1];
    float d2 = 1.0f;
    float dir[3] = {d0, d1, d2};
    row[256] = (_Float16)d0;
    row[257] = (_Float16)d1;
    row[258] = (_Float16)d2;
    float f = 1.0f;
    for (int fr = 0; fr < 4; ++fr) {
        for (int d = 0; d < 3; ++d) {
            float v = dir[d] * f;
            row[259 + fr * 3 + d] = (_Float16)__sinf(v);
            row[271 + fr * 3 + d] = (_Float16)__cosf(v);
        }
        f *= 2.0f;
    }
    for (int i = 283; i < 320; ++i) row[i] = (_Float16)0.0f;
}

// ---------------------------------------------------------------------------
// Composite: out = sum(sigmoid(raw)*w) + bg*(1 - sum(w))
// ---------------------------------------------------------------------------
__global__ void k_composite(const _Float16* __restrict__ raw, const float* __restrict__ wbuf,
                            const float* __restrict__ bg, float* __restrict__ out, int n, int B) {
    int ray = blockIdx.x * blockDim.x + threadIdx.x;
    if (ray >= B) return;
    float acc0 = 0.0f, acc1 = 0.0f, acc2 = 0.0f, wsum = 0.0f;
    for (int s = 0; s < n; ++s) {
        float w = wbuf[(size_t)ray * n + s];
        wsum += w;
        const _Float16* c = raw + ((size_t)ray * n + s) * 16;
        acc0 += w / (1.0f + __expf(-(float)c[0]));
        acc1 += w / (1.0f + __expf(-(float)c[1]));
        acc2 += w / (1.0f + __expf(-(float)c[2]));
    }
    float rem = 1.0f - wsum;
    out[ray * 3 + 0] = acc0 + bg[ray * 3 + 0] * rem;
    out[ray * 3 + 1] = acc1 + bg[ray * 3 + 1] * rem;
    out[ray * 3 + 2] = acc2 + bg[ray * 3 + 2] * rem;
}

// ---------------------------------------------------------------------------
// Host orchestration
// ---------------------------------------------------------------------------
extern "C" void kernel_launch(void* const* d_in, const int* in_sizes, int n_in,
                              void* d_out, int out_size, void* d_ws, size_t ws_size,
                              hipStream_t stream) {
    (void)in_sizes; (void)n_in; (void)out_size; (void)ws_size;
    const int B = 2048, NS = 64, NF = 128;
    const int Mc = B * NS;   // 131072 coarse points
    const int Mf = B * NF;   // 262144 fine points

    const float* coords = (const float*)d_in[0];
    // geo_coarse: W,b pairs d_in[1..8]; color_coarse d_in[9..14] (dead code);
    // geo_fine d_in[15..22]; color_fine d_in[23..28]
    const float* dW = (const float*)d_in[29];
    const float* db = (const float*)d_in[30];
    const float* noise   = (const float*)d_in[31];
    const float* u_fine  = (const float*)d_in[32];
    const float* bg_fine = (const float*)d_in[34];

    char* base = (char*)d_ws;
    size_t off = 0;
    auto alloc = [&](size_t bytes) -> char* {
        char* p = base + off;
        off = (off + bytes + 255) & ~(size_t)255;
        return p;
    };

    _Float16* wt_gc[4]; _Float16* wt_gf[4]; _Float16* wt_cf[3];
    const int geoK[4]  = {63, 256, 256, 256}, geoKp[4] = {64, 256, 256, 256};
    const int geoN[4]  = {256, 256, 256, 512};
    const int colK[3]  = {283, 128, 128},     colKp[3] = {320, 128, 128};
    const int colN[3]  = {128, 128, 3},       colNp[3] = {128, 128, 16};
    for (int i = 0; i < 4; ++i) wt_gc[i] = (_Float16*)alloc((size_t)geoN[i] * geoKp[i] * 2);
    for (int i = 0; i < 4; ++i) wt_gf[i] = (_Float16*)alloc((size_t)geoN[i] * geoKp[i] * 2);
    for (int i = 0; i < 3; ++i) wt_cf[i] = (_Float16*)alloc((size_t)colNp[i] * colKp[i] * 2);

    _Float16* bufA    = (_Float16*)alloc((size_t)Mf * 320 * 2); // geo in (x64) / color in (x320)
    _Float16* bufB    = (_Float16*)alloc((size_t)Mf * 256 * 2);
    _Float16* bufC    = (_Float16*)alloc((size_t)Mf * 256 * 2);
    _Float16* bufFeat = (_Float16*)alloc((size_t)Mf * 512 * 2);
    float* dens    = (float*)alloc((size_t)Mf * 4);
    float* wbuf    = (float*)alloc((size_t)B * 128 * 4);
    float* steps_c = (float*)alloc((size_t)B * 64 * 4);
    float* bound_c = (float*)alloc((size_t)B * 65 * 4);
    float* steps_f = (float*)alloc((size_t)B * 128 * 4);
    float* bound_f = (float*)alloc((size_t)B * 129 * 4);

    auto prep = [&](const float* W, _Float16* WT, int K, int N, int Kp, int Np) {
        int total = Kp * Np;
        k_prep_weight<<<(total + 255) / 256, 256, 0, stream>>>(W, WT, K, N, Kp, Np);
    };
    for (int i = 0; i < 4; ++i) prep((const float*)d_in[1 + 2 * i],  wt_gc[i], geoK[i], geoN[i], geoKp[i], geoN[i]);
    for (int i = 0; i < 4; ++i) prep((const float*)d_in[15 + 2 * i], wt_gf[i], geoK[i], geoN[i], geoKp[i], geoN[i]);
    for (int i = 0; i < 3; ++i) prep((const float*)d_in[23 + 2 * i], wt_cf[i], colK[i], colN[i], colKp[i], colNp[i]);

    // ---------------- Coarse pass (geo + density only; color MLP is dead) ----
    k_steps_coarse<<<(B + 255) / 256, 256, 0, stream>>>(noise, steps_c, bound_c, B);
    k_embed_geo<<<(Mc + 255) / 256, 256, 0, stream>>>(coords, steps_c, bufA, NS, Mc);

    gemm_wmma<8, true ><<<dim3(Mc / 64, 2), 128, 0, stream>>>(bufA, wt_gc[0], (const float*)d_in[2], bufB, 64, 256, 256);
    gemm_wmma<8, true ><<<dim3(Mc / 64, 2), 128, 0, stream>>>(bufB, wt_gc[1], (const float*)d_in[4], bufC, 256, 256, 256);
    gemm_wmma<8, true ><<<dim3(Mc / 64, 2), 128, 0, stream>>>(bufC, wt_gc[2], (const float*)d_in[6], bufB, 256, 256, 256);
    gemm_wmma<8, false><<<dim3(Mc / 64, 4), 128, 0, stream>>>(bufB, wt_gc[3], (const float*)d_in[8], bufFeat, 256, 512, 512);

    k_density<<<(Mc + 255) / 256, 256, 0, stream>>>(bufFeat, dW, db, dens, Mc);
    k_weights<<<(B + 255) / 256, 256, 0, stream>>>(dens, bound_c, wbuf, NS, B);

    // ---------------- Importance sampling + sort ----------------------------
    k_sample_sort<<<B, 64, 0, stream>>>(steps_c, bound_c, wbuf, u_fine, steps_f, bound_f);

    // ---------------- Fine pass ---------------------------------------------
    k_embed_geo<<<(Mf + 255) / 256, 256, 0, stream>>>(coords, steps_f, bufA, NF, Mf);

    gemm_wmma<8, true ><<<dim3(Mf / 64, 2), 128, 0, stream>>>(bufA, wt_gf[0], (const float*)d_in[16], bufB, 64, 256, 256);
    gemm_wmma<8, true ><<<dim3(Mf / 64, 2), 128, 0, stream>>>(bufB, wt_gf[1], (const float*)d_in[18], bufC, 256, 256, 256);
    gemm_wmma<8, true ><<<dim3(Mf / 64, 2), 128, 0, stream>>>(bufC, wt_gf[2], (const float*)d_in[20], bufB, 256, 256, 256);
    gemm_wmma<8, false><<<dim3(Mf / 64, 4), 128, 0, stream>>>(bufB, wt_gf[3], (const float*)d_in[22], bufFeat, 256, 512, 512);

    k_density<<<(Mf + 255) / 256, 256, 0, stream>>>(bufFeat, dW, db, dens, Mf);
    k_weights<<<(B + 255) / 256, 256, 0, stream>>>(dens, bound_f, wbuf, NF, B);

    k_embed_col<<<(Mf + 255) / 256, 256, 0, stream>>>(bufFeat, coords, bufA, NF, Mf);

    gemm_wmma<8, true ><<<dim3(Mf / 64, 1), 128, 0, stream>>>(bufA, wt_cf[0], (const float*)d_in[24], bufB, 320, 128, 128);
    gemm_wmma<8, true ><<<dim3(Mf / 64, 1), 128, 0, stream>>>(bufB, wt_cf[1], (const float*)d_in[26], bufC, 128, 128, 128);
    gemm_wmma<1, false><<<dim3(Mf / 64, 1), 128, 0, stream>>>(bufC, wt_cf[2], (const float*)d_in[28], bufB, 128, 16, 3);

    k_composite<<<(B + 255) / 256, 256, 0, stream>>>(bufB, wbuf, bg_fine, (float*)d_out, NF, B);
}
